// Jin_2310692405381
// MI455X (gfx1250) — matrix operand order repaired and checked by reading, not compiled, not run on hardware
//
#include <hip/hip_runtime.h>
#include <math.h>

// ---------------- problem constants (match reference) ----------------
#define NND   50000     // nodes
#define FIN   3000      // input features
#define KP1   3008      // FIN padded to multiple of 32 for WMMA K
#define HID   512
#define LAT   256
#define EREM  480000
#define EMSK  1120000

// ---------------- WMMA fragment types (gfx1250, wave32) ----------------
typedef __attribute__((ext_vector_type(16))) __bf16 bf16x16;
typedef __attribute__((ext_vector_type(8)))  __bf16 bf16x8;
typedef __attribute__((ext_vector_type(8)))  float  f32x8;

// A fragment: 16x32 bf16, row-major A with leading dim lda.
// Lane l: m = tileRow + (l&15); g = l>>4.
// elems 0..7 = K[k0+8g .. +7], elems 8..15 = K[k0+8g+16 .. +23]   (ISA 7.12.2)
__device__ __forceinline__ bf16x16 load_a_frag(const __bf16* A, size_t m,
                                               int lda, int k0, int g) {
  union { bf16x16 v; bf16x8 h[2]; } u;
  const __bf16* base = A + m * (size_t)lda + k0 + 8 * g;
  u.h[0] = *(const bf16x8*)(base);
  u.h[1] = *(const bf16x8*)(base + 16);
  return u.v;
}

// B fragment from TRANSPOSED weights BT[N x K] row-major:
// Lane l: n = tileCol + (l&15); g = l>>4; elems 0..7 -> K[k0+8g .. +7], elems 8..15 -> K[k0+8g+16 .. +23].
__device__ __forceinline__ bf16x16 load_b_frag(const __bf16* BT, size_t n,
                                               int ldk, int k0, int g) {
  union { bf16x16 v; bf16x8 h[2]; } u;
  const __bf16* base = BT + n * (size_t)ldk + k0 + 8 * g;
  u.h[0] = *(const bf16x8*)(base);
  u.h[1] = *(const bf16x8*)(base + 16);
  return u.v;
}

// B fragment from an LDS panel with padded row stride (in bf16 elements).
__device__ __forceinline__ bf16x16 lds_b_frag(const __bf16* P, int ln, int g,
                                              int ks, int stride) {
  union { bf16x16 v; bf16x8 h[2]; } u;
  const __bf16* base = P + ln * stride + ks * 32 + 8 * g;
  u.h[0] = *(const bf16x8*)(base);
  u.h[1] = *(const bf16x8*)(base + 16);
  return u.v;
}

// ---------------- async copy helpers (gfx1250 ASYNCcnt path) ----------------
__device__ __forceinline__ void async_b128(const void* gbase, unsigned goff,
                                           void* lds_generic) {
  // GVS mode: mem = SGPR64 + VGPR32 + ioffset. LDS dest VGPR = low 32 bits of
  // the generic pointer (= LDS byte offset). Tracked by ASYNCcnt.
  unsigned lds = (unsigned)(unsigned long long)lds_generic;
  asm volatile("global_load_async_to_lds_b128 %0, %1, %2"
               :: "v"(lds), "v"(goff), "s"((unsigned long long)gbase)
               : "memory");
}
__device__ __forceinline__ void wait_asynccnt0() {
#if __has_builtin(__builtin_amdgcn_s_wait_asynccnt)
  __builtin_amdgcn_s_wait_asynccnt(0);
#else
  asm volatile("s_wait_asynccnt 0x0" ::: "memory");
#endif
}

// ---------------- conversion / padding kernels ----------------
__global__ void cvt_pad_bf16(const float* __restrict__ X, __bf16* __restrict__ Xb,
                             int M, int K, int Kpad) {
  size_t i = blockIdx.x * (size_t)blockDim.x + threadIdx.x;
  if (i >= (size_t)M * Kpad) return;
  int k = (int)(i % Kpad);
  size_t m = i / Kpad;
  Xb[i] = (k < K) ? (__bf16)X[m * (size_t)K + k] : (__bf16)0.f;
}

// W[K x N] (row major) -> WT[N x Kpad] bf16 (zero pad K tail)
__global__ void cvt_wT_bf16(const float* __restrict__ W, __bf16* __restrict__ WT,
                            int K, int N, int Kpad) {
  size_t i = blockIdx.x * (size_t)blockDim.x + threadIdx.x;
  if (i >= (size_t)N * Kpad) return;
  int k = (int)(i % Kpad);
  size_t n = i / Kpad;
  WT[i] = (k < K) ? (__bf16)W[(size_t)k * N + n] : (__bf16)0.f;
}

// ---------------- WMMA GEMM: C[MxN] = A[MxKpad](bf16) * BT[NxKpad](bf16) ------
// Register-blocked: one wave -> 16x64 C tile (1 A-frag reused across 4 B-frags,
// 4 f32x8 accumulators) => 1.25 fragment loads per WMMA instead of 2.
__global__ void wmma_gemm_bf16(const __bf16* __restrict__ A,
                               const __bf16* __restrict__ BT,
                               float* __restrict__ C,
                               int M, int N, int Kpad) {
  const int lane = threadIdx.x & 31;
  const int wave = threadIdx.x >> 5;
  const int nb = N >> 6;                       // 64-wide N blocks
  const int job = blockIdx.x * (blockDim.x >> 5) + wave;
  if (job >= (M >> 4) * nb) return;            // wave-uniform
  const int tm = job / nb, tnb = job % nb;
  const int g = lane >> 4, ln = lane & 15;
  const size_t am = (size_t)(tm * 16 + ln);

  f32x8 acc0 = {}, acc1 = {}, acc2 = {}, acc3 = {};
  for (int k0 = 0; k0 < Kpad; k0 += 32) {
    bf16x16 a = load_a_frag(A, am, Kpad, k0, g);
    bf16x16 b0 = load_b_frag(BT, (size_t)(tnb * 64 + 0  + ln), Kpad, k0, g);
    bf16x16 b1 = load_b_frag(BT, (size_t)(tnb * 64 + 16 + ln), Kpad, k0, g);
    bf16x16 b2 = load_b_frag(BT, (size_t)(tnb * 64 + 32 + ln), Kpad, k0, g);
    bf16x16 b3 = load_b_frag(BT, (size_t)(tnb * 64 + 48 + ln), Kpad, k0, g);
    acc0 = __builtin_amdgcn_wmma_f32_16x16x32_bf16(false, a, false, b0, (short)0, acc0, false, false);
    acc1 = __builtin_amdgcn_wmma_f32_16x16x32_bf16(false, a, false, b1, (short)0, acc1, false, false);
    acc2 = __builtin_amdgcn_wmma_f32_16x16x32_bf16(false, a, false, b2, (short)0, acc2, false, false);
    acc3 = __builtin_amdgcn_wmma_f32_16x16x32_bf16(false, a, false, b3, (short)0, acc3, false, false);
  }
  // C layout: VGPR r -> row tm*16 + r + 8g, col = subtile*16 + ln
  float* crow = C + (size_t)(tnb * 64 + ln);
#pragma unroll
  for (int r = 0; r < 8; ++r) {
    size_t row = (size_t)(tm * 16 + r + 8 * g) * N;
    crow[row]      = acc0[r];
    crow[row + 16] = acc1[r];
    crow[row + 32] = acc2[r];
    crow[row + 48] = acc3[r];
  }
}

// ---------------- GCN normalization / aggregation ----------------
__global__ void deg_init(float* deg, int n) {     // self-loop folded in
  int i = blockIdx.x * blockDim.x + threadIdx.x;
  if (i < n) deg[i] = 1.f;
}
__global__ void deg_scatter(const int* __restrict__ col, float* deg, int E) {
  int i = blockIdx.x * blockDim.x + threadIdx.x;
  if (i < E) atomicAdd(&deg[col[i]], 1.f);
}
__global__ void dinv_kernel(const float* __restrict__ deg, float* dinv, int n) {
  int i = blockIdx.x * blockDim.x + threadIdx.x;
  if (i < n) dinv[i] = (deg[i] > 0.f) ? rsqrtf(deg[i]) : 0.f;
}
__global__ void agg_init(const float* __restrict__ H, const float* __restrict__ dinv,
                         float* __restrict__ Agg, int F, size_t total) {
  size_t i = blockIdx.x * (size_t)blockDim.x + threadIdx.x;
  if (i >= total) return;
  float d = dinv[i / F];
  Agg[i] = H[i] * d * d;
}
__global__ void agg_scatter(const int* __restrict__ row, const int* __restrict__ col,
                            const float* __restrict__ H, const float* __restrict__ dinv,
                            float* __restrict__ Agg, int E, int F) {
  int chunks = F >> 3;
  size_t t = blockIdx.x * (size_t)blockDim.x + threadIdx.x;
  if (t >= (size_t)E * chunks) return;
  int e = (int)(t / chunks), c = (int)(t % chunks);
  int r = row[e], cl = col[e];
  float nrm = dinv[r] * dinv[cl];
  const float* hs = H + (size_t)r * F + c * 8;
  float* dst = Agg + (size_t)cl * F + c * 8;
#pragma unroll
  for (int j = 0; j < 8; ++j) atomicAdd(dst + j, hs[j] * nrm);
}
__global__ void bias_elu(float* __restrict__ Agg, const float* __restrict__ b,
                         __bf16* __restrict__ Zb, int F, size_t total) {
  size_t i = blockIdx.x * (size_t)blockDim.x + threadIdx.x;
  if (i >= total) return;
  float v = Agg[i] + b[i % F];
  v = (v > 0.f) ? v : expm1f(v);
  Agg[i] = v;
  if (Zb) Zb[i] = (__bf16)v;
}

// ---------------- fused edge decoder with async LDS B-panel staging ----------
// Block = 8 waves x 16 edges = 128 edges. Per n-tile (16 cols of fc1), the block
// DMAs the 16x256 bf16 eW1^T panel into LDS ONCE (global_load_async_to_lds_b128,
// double-buffered, overlapped with WMMA) instead of 8 waves re-reading it.
// LDS rows padded to 264 bf16 (528 B): (528/4) mod 64 = 4 -> 16 fragment lanes
// hit 16 distinct banks.
#define PSTRIDE 264           // bf16 elements per padded LDS row
__global__ void edge_decoder(const float* __restrict__ Z,     // [NND x LAT] f32
                             const int* __restrict__ e0, const int* __restrict__ e1,
                             const __bf16* __restrict__ eW1T, // [HID x LAT] bf16
                             const float* __restrict__ eb1,   // [HID]
                             const float* __restrict__ eW2,   // [HID]
                             const float* __restrict__ eb2,   // [1]
                             float* __restrict__ logits, int E) {
  __shared__ __bf16 panel[2][16 * PSTRIDE];    // 2 x 8448 B
  const int lane = threadIdx.x & 31;
  const int wave = threadIdx.x >> 5;
  const int t    = threadIdx.x;
  if ((int)blockIdx.x * 128 >= E) return;      // block-uniform (never with exact grid)
  const int g = lane >> 4, ln = lane & 15;
  const int tile = blockIdx.x * 8 + wave;
  const int edge = tile * 16 + ln;             // A row m = ln for this lane
  const float* z0 = Z + (size_t)e0[edge] * LAT;
  const float* z1 = Z + (size_t)e1[edge] * LAT;

  // A[m,k] = z0[k]*z1[k] in bf16, all 8 K-step fragments cached in registers.
  bf16x16 afrag[8];
#pragma unroll
  for (int ks = 0; ks < 8; ++ks) {
    int kb = ks * 32 + 8 * g;
    union { bf16x16 v; __bf16 e[16]; } u;
#pragma unroll
    for (int j = 0; j < 8; ++j) {
      u.e[j]     = (__bf16)(z0[kb + j]      * z1[kb + j]);
      u.e[8 + j] = (__bf16)(z0[kb + 16 + j] * z1[kb + 16 + j]);
    }
    afrag[ks] = u.v;
  }

  // Async-stage one 16x256 panel: 512 chunks of 16 B; 256 threads x 2 chunks.
  auto load_panel = [&](int tn, int buf) {
#pragma unroll
    for (int i = 0; i < 2; ++i) {
      int c   = t + i * 256;
      int row = c >> 5, q = c & 31;            // 32 chunks per 512 B row
      unsigned goff = (unsigned)(((tn * 16 + row) * LAT + q * 8) * 2);
      async_b128(eW1T, goff, &panel[buf][row * PSTRIDE + q * 8]);
    }
  };

  load_panel(0, 0);                            // prologue DMA
  float partial[8] = {0.f, 0.f, 0.f, 0.f, 0.f, 0.f, 0.f, 0.f};
  for (int tn = 0; tn < HID / 16; ++tn) {
    wait_asynccnt0();                          // this wave's chunks landed
    __syncthreads();                           // whole panel visible; prev compute done
    if (tn + 1 < HID / 16) load_panel(tn + 1, (tn + 1) & 1);  // overlap next DMA
    const __bf16* P = panel[tn & 1];
    f32x8 acc = {};
#pragma unroll
    for (int ks = 0; ks < 8; ++ks) {
      bf16x16 b = lds_b_frag(P, ln, g, ks, PSTRIDE);
      acc = __builtin_amdgcn_wmma_f32_16x16x32_bf16(false, afrag[ks], false, b,
                                                    (short)0, acc, false, false);
    }
    const int n = tn * 16 + ln;
    const float bb = eb1[n], w2 = eW2[n];
#pragma unroll
    for (int r = 0; r < 8; ++r) {
      float h = acc[r] + bb;
      partial[r] += (h > 0.f ? h : 0.f) * w2;  // ReLU then fc2 dot
    }
  }
  // reduce over the 16-lane n dimension (xor of bits 0..3 stays in half-wave)
#pragma unroll
  for (int r = 0; r < 8; ++r) {
    float v = partial[r];
    for (int m = 1; m < 16; m <<= 1) v += __shfl_xor(v, m, 32);
    if (ln == 0) logits[tile * 16 + 8 * g + r] = v + eb2[0];  // row m = r + 8g
  }
}

// ---------------- InfoNCE ----------------
__global__ void zero1(float* p) { if (threadIdx.x == 0 && blockIdx.x == 0) p[0] = 0.f; }

__global__ void infonce_loss(const float* __restrict__ pos, const float* __restrict__ neg,
                             float* __restrict__ acc, int E) {
  __shared__ float sdata[256];
  float s = 0.f;
  for (int i = blockIdx.x * blockDim.x + threadIdx.x; i < E;
       i += gridDim.x * blockDim.x) {
    float pe = expf(pos[i]), ne = expf(neg[i]);
    s += -logf(pe / (pe + ne) + 1e-15f);
  }
  sdata[threadIdx.x] = s;
  __syncthreads();
  for (int st = 128; st > 0; st >>= 1) {
    if (threadIdx.x < st) sdata[threadIdx.x] += sdata[threadIdx.x + st];
    __syncthreads();
  }
  if (threadIdx.x == 0) atomicAdd(acc, sdata[0]);
}
__global__ void finalize(const float* acc, float* out, float inv) {
  if (threadIdx.x == 0 && blockIdx.x == 0) out[0] = acc[0] * inv;
}

// ---------------- host-side orchestration ----------------
static inline unsigned blocks_for(size_t total, int bs) {
  return (unsigned)((total + bs - 1) / bs);
}

extern "C" void kernel_launch(void* const* d_in, const int* in_sizes, int n_in,
                              void* d_out, int out_size, void* d_ws, size_t ws_size,
                              hipStream_t stream) {
  const float* x   = (const float*)d_in[0];
  const int*   rem = (const int*)d_in[1];     // [2, EREM]
  const int*   mske= (const int*)d_in[2];     // [2, EMSK]
  const int*   nege= (const int*)d_in[3];     // [2, EMSK]
  const float* W1  = (const float*)d_in[4];
  const float* b1  = (const float*)d_in[5];
  const float* W2  = (const float*)d_in[6];
  const float* b2  = (const float*)d_in[7];
  const float* eW1 = (const float*)d_in[8];
  const float* eb1 = (const float*)d_in[9];
  const float* eW2 = (const float*)d_in[10];
  const float* eb2 = (const float*)d_in[11];
  float* out = (float*)d_out;

  // workspace bump allocator (256B aligned)
  char* w = (char*)d_ws;
  size_t off = 0;
  auto alloc = [&](size_t bytes) {
    void* p = w + off;
    off += (bytes + 255) & ~(size_t)255;
    return p;
  };
  __bf16* Xb   = (__bf16*)alloc((size_t)NND * KP1 * 2);   // 300.8 MB
  __bf16* W1T  = (__bf16*)alloc((size_t)HID * KP1 * 2);
  float*  H1   = (float*) alloc((size_t)NND * HID * 4);
  float*  Agg1 = (float*) alloc((size_t)NND * HID * 4);   // becomes z1 (f32)
  __bf16* Z1b  = (__bf16*)alloc((size_t)NND * HID * 2);
  __bf16* W2T  = (__bf16*)alloc((size_t)LAT * HID * 2);
  float*  H2   = (float*) alloc((size_t)NND * LAT * 4);
  float*  Agg2 = (float*) alloc((size_t)NND * LAT * 4);   // becomes z (latent)
  __bf16* eW1T = (__bf16*)alloc((size_t)HID * LAT * 2);
  float*  deg  = (float*) alloc((size_t)NND * 4);
  float*  dinv = (float*) alloc((size_t)NND * 4);
  float*  posL = (float*) alloc((size_t)EMSK * 4);
  float*  negL = (float*) alloc((size_t)EMSK * 4);
  float*  lossA= (float*) alloc(4);
  (void)ws_size; (void)n_in; (void)in_sizes; (void)out_size;

  const int BS = 256;

  // 1) conversions: x -> bf16 padded; weights -> transposed bf16
  cvt_pad_bf16<<<blocks_for((size_t)NND * KP1, BS), BS, 0, stream>>>(x, Xb, NND, FIN, KP1);
  cvt_wT_bf16<<<blocks_for((size_t)HID * KP1, BS), BS, 0, stream>>>(W1, W1T, FIN, HID, KP1);
  cvt_wT_bf16<<<blocks_for((size_t)LAT * HID, BS), BS, 0, stream>>>(W2, W2T, HID, LAT, HID);
  cvt_wT_bf16<<<blocks_for((size_t)HID * LAT, BS), BS, 0, stream>>>(eW1, eW1T, LAT, HID, LAT);

  // 2) degree / rsqrt norm (self-loops folded into init value 1)
  deg_init<<<blocks_for(NND, BS), BS, 0, stream>>>(deg, NND);
  deg_scatter<<<blocks_for(EREM, BS), BS, 0, stream>>>(rem + EREM, deg, EREM);
  dinv_kernel<<<blocks_for(NND, BS), BS, 0, stream>>>(deg, dinv, NND);

  // 3) layer 1: H1 = Xb @ W1 -> aggregate -> +b1, ELU, bf16
  {
    unsigned jobs = (NND / 16) * (HID / 64);               // 25000
    wmma_gemm_bf16<<<blocks_for(jobs, 8), BS, 0, stream>>>(Xb, W1T, H1, NND, HID, KP1);
  }
  agg_init<<<blocks_for((size_t)NND * HID, BS), BS, 0, stream>>>(H1, dinv, Agg1, HID, (size_t)NND * HID);
  agg_scatter<<<blocks_for((size_t)EREM * (HID / 8), BS), BS, 0, stream>>>(
      rem, rem + EREM, H1, dinv, Agg1, EREM, HID);
  bias_elu<<<blocks_for((size_t)NND * HID, BS), BS, 0, stream>>>(Agg1, b1, Z1b, HID, (size_t)NND * HID);

  // 4) layer 2: H2 = z1 @ W2 -> aggregate -> +b2, ELU -> latent z (in Agg2)
  {
    unsigned jobs = (NND / 16) * (LAT / 64);               // 12500
    wmma_gemm_bf16<<<blocks_for(jobs, 8), BS, 0, stream>>>(Z1b, W2T, H2, NND, LAT, HID);
  }
  agg_init<<<blocks_for((size_t)NND * LAT, BS), BS, 0, stream>>>(H2, dinv, Agg2, LAT, (size_t)NND * LAT);
  agg_scatter<<<blocks_for((size_t)EREM * (LAT / 8), BS), BS, 0, stream>>>(
      rem, rem + EREM, H2, dinv, Agg2, EREM, LAT);
  bias_elu<<<blocks_for((size_t)NND * LAT, BS), BS, 0, stream>>>(Agg2, b2, (__bf16*)nullptr, LAT, (size_t)NND * LAT);

  // 5) fused edge decoder (pos + neg), async-staged B panels
  {
    unsigned blocksE = EMSK / 128;                          // 8750 exact
    edge_decoder<<<blocksE, BS, 0, stream>>>(Agg2, mske, mske + EMSK, eW1T, eb1, eW2, eb2, posL, EMSK);
    edge_decoder<<<blocksE, BS, 0, stream>>>(Agg2, nege, nege + EMSK, eW1T, eb1, eW2, eb2, negL, EMSK);
  }

  // 6) InfoNCE
  zero1<<<1, 64, 0, stream>>>(lossA);
  infonce_loss<<<1024, BS, 0, stream>>>(posL, negL, lossA, EMSK);
  finalize<<<1, 64, 0, stream>>>(lossA, out, 1.0f / (float)EMSK);
}
